// implicit_pointnet_58566174049040
// MI455X (gfx1250) — compile-verified, matrix operand order checked by reading, not run on hardware
//
#include <hip/hip_runtime.h>
#include <hip/hip_bf16.h>
#include <math.h>

// ---------------------------------------------------------------------------
// implicit pointnet pipeline for MI455X (gfx1250, wave32, WMMA + TDM)
//   outputs (f32, concatenated): Proj[16,1,128,128], V[16,128,128,128],
//                                posn[16,16384,3], Wb[16,1,96,96], res[16,16384,3]
// ---------------------------------------------------------------------------

typedef __attribute__((ext_vector_type(16))) _Float16     v16h;
typedef __attribute__((ext_vector_type(8)))  float        v8f;
typedef __attribute__((ext_vector_type(4)))  unsigned int v4u;
typedef __attribute__((ext_vector_type(8)))  int          v8i;
typedef __attribute__((ext_vector_type(4)))  int          v4i;

#ifndef HAVE_TDM
#if defined(__has_builtin)
#if __has_builtin(__builtin_amdgcn_tensor_load_to_lds) && \
    __has_builtin(__builtin_amdgcn_s_wait_tensorcnt)
#define HAVE_TDM 1
#endif
#endif
#endif
#ifndef HAVE_TDM
#define HAVE_TDM 0
#endif

#define BOX    128
#define NPTS   16384
#define BATCH  16
#define VOXB   (BOX * BOX * BOX)            // 2097152

// ---- d_out offsets (in floats) --------------------------------------------
#define O_PROJ 0L
#define N_PROJ ((long)BATCH * BOX * BOX)    // 262144
#define O_VOL  (O_PROJ + N_PROJ)
#define N_VOL  ((long)BATCH * VOXB)         // 33554432
#define O_POSN (O_VOL + N_VOL)
#define N_POSN ((long)BATCH * NPTS * 3)
#define O_WB   (O_POSN + N_POSN)
#define N_WB   ((long)BATCH * 96 * 96)
#define O_RES  (O_WB + N_WB)

// ---- workspace offsets (bytes); total requirement ~302 MB -----------------
#define WS_A16F  0L           // f16 [19][32][16]        A fragments (19456 B)
#define WS_RES0  32768L       // f32 [16][6144]
#define WS_C0    458752L      // f32 [16][6][4096]
#define WS_C1    2097152L     // f32 [16][3][8192]
#define WS_C2    4194304L     // f32 [16][3][16384]
#define WS_SLICE 8388608L     // float2 [16][128][128]
#define WS_W6F   10485760L    // f16 [384][19][32][16]   B fragments (7471104 B)
#define WS_VC    33554432L    // float2 [16][128][128][128]  (268 MB)

#define NKT   19              // ceil(600/32)
#define NTILE 384             // 6144/16

// ===========================================================================
// 1. Small MLP layers 1..5 (scalar, one workgroup) + A-fragment emission.
//    f16 A/B fragment layout (16x32): lane L -> half=L>>4, m/n=L&15,
//    element e -> K = e + 8*((e>>3)+half).
// ===========================================================================
__global__ __launch_bounds__(256)
void mlp_head(const float* __restrict__ z,  const float* __restrict__ w1,
              const float* __restrict__ w2, const float* __restrict__ w3,
              const float* __restrict__ w4, const float* __restrict__ w5,
              _Float16* __restrict__ a16f) {
  __shared__ float actA[BATCH * 150];
  __shared__ float actB[BATCH * 150];
  __shared__ float act5[BATCH * 608];
  const int t = threadIdx.x;

  // L1: [16,8] @ [8,10] -> relu -> actA
  for (int idx = t; idx < BATCH * 10; idx += 256) {
    int b = idx / 10, n = idx % 10;
    float s = 0.f;
    for (int k = 0; k < 8; ++k) s = fmaf(z[b * 8 + k], w1[k * 10 + n], s);
    actA[b * 10 + n] = fmaxf(s, 0.f);
  }
  __syncthreads();
  // L2: [16,10] @ [10,10] -> relu -> actB
  for (int idx = t; idx < BATCH * 10; idx += 256) {
    int b = idx / 10, n = idx % 10;
    float s = 0.f;
    for (int k = 0; k < 10; ++k) s = fmaf(actA[b * 10 + k], w2[k * 10 + n], s);
    actB[b * 10 + n] = fmaxf(s, 0.f);
  }
  __syncthreads();
  // L3: [16,10] @ [10,30] -> relu -> actA
  for (int idx = t; idx < BATCH * 30; idx += 256) {
    int b = idx / 30, n = idx % 30;
    float s = 0.f;
    for (int k = 0; k < 10; ++k) s = fmaf(actB[b * 10 + k], w3[k * 30 + n], s);
    actA[b * 30 + n] = fmaxf(s, 0.f);
  }
  __syncthreads();
  // L4: [16,30] @ [30,150] -> relu -> actB
  for (int idx = t; idx < BATCH * 150; idx += 256) {
    int b = idx / 150, n = idx % 150;
    float s = 0.f;
    for (int k = 0; k < 30; ++k) s = fmaf(actA[b * 30 + k], w4[k * 150 + n], s);
    actB[b * 150 + n] = fmaxf(s, 0.f);
  }
  __syncthreads();
  // L5: [16,150] @ [150,600] -> relu, zero-padded to K=608
  for (int idx = t; idx < BATCH * 608; idx += 256) {
    int b = idx / 608, n = idx % 608;
    float s = 0.f;
    if (n < 600) {
      for (int k = 0; k < 150; ++k) s = fmaf(actB[b * 150 + k], w5[k * 600 + n], s);
      s = fmaxf(s, 0.f);
    }
    act5[b * 608 + n] = s;
  }
  __syncthreads();
  // Emit WMMA A-fragments: a16f[kt][lane][e] = act5[m][kt*32 + K(e,half)]
  for (int idx = t; idx < NKT * 512; idx += 256) {
    int e = idx & 15, lane = (idx >> 4) & 31, kt = idx >> 9;
    int half = lane >> 4, m = lane & 15;
    int K = e + 8 * ((e >> 3) + half);
    a16f[idx] = (_Float16)act5[m * 608 + kt * 32 + K];
  }
}

// ===========================================================================
// 2a. Pack w6[600,6144] (f32) into f16 WMMA B-fragments, K padded to 608.
//     w6f[tile][kt][lane][e] = w6[kt*32 + K(e,half)][tile*16 + n]
// ===========================================================================
__global__ void w6_to_frag(const float* __restrict__ w6, _Float16* __restrict__ w6f) {
  long idx = (long)blockIdx.x * blockDim.x + threadIdx.x;
  if (idx >= (long)NTILE * NKT * 512) return;
  int e    = idx & 15;
  int lane = (int)((idx >> 4) & 31);
  long r   = idx >> 9;
  int kt   = (int)(r % NKT);
  int tile = (int)(r / NKT);
  int half = lane >> 4, n = lane & 15;
  int K = e + 8 * ((e >> 3) + half);
  int k = kt * 32 + K;
  w6f[idx] = (k < 600) ? (_Float16)w6[(long)k * 6144 + tile * 16 + n]
                       : (_Float16)0.f;
}

// ===========================================================================
// 2b. GEMM res0[16,6144] = A[16,600] @ w6[600,6144] via v_wmma_f32_16x16x32_f16
//     One wave per 16-col tile; fully unrolled 19-step K loop; contiguous
//     32 B fragment loads per lane.  f32 C/D: row m = v + 8*(L>>4), col = L&15.
// ===========================================================================
__global__ __launch_bounds__(128)
void gemm_w6_wmma(const _Float16* __restrict__ a16f,
                  const _Float16* __restrict__ w6f,
                  float* __restrict__ res0) {
  const int lane = threadIdx.x & 31;
  const int wave = threadIdx.x >> 5;
  const int tile = blockIdx.x * 4 + wave;          // 96 blocks * 4 waves = 384
  const _Float16* ap = a16f + lane * 16;
  const _Float16* bp = w6f + (long)tile * (NKT * 512) + lane * 16;

  v8f c = {};
#pragma unroll
  for (int kt = 0; kt < NKT; ++kt) {
    v16h a = *(const v16h*)(ap + kt * 512);
    v16h b = *(const v16h*)(bp + kt * 512);
    c = __builtin_amdgcn_wmma_f32_16x16x32_f16(false, a, false, b,
                                               (short)0, c, false, false);
  }
  const int half = lane >> 4, n = lane & 15, col0 = tile * 16;
#pragma unroll
  for (int v = 0; v < 8; ++v)
    res0[(long)(v + 8 * half) * 6144 + col0 + n] = c[v];
}

// ===========================================================================
// 3. Upsample(x2) + conv1d chain (fused repeat)
// ===========================================================================
__global__ void convA_kernel(const float* __restrict__ res0,     // [16,3,2048]
                             const float* __restrict__ c0,       // [6,3,3]
                             float* __restrict__ outA) {         // [16,6,4096]
  long idx = (long)blockIdx.x * blockDim.x + threadIdx.x;
  if (idx >= (long)BATCH * 6 * 4096) return;
  int l = idx % 4096; int o = (idx / 4096) % 6; int b = idx / (4096 * 6);
  float s = 0.f;
  for (int i = 0; i < 3; ++i) {
    const float* xin = res0 + (long)b * 6144 + i * 2048;
    for (int tt = 0; tt < 3; ++tt) {
      int m = l + tt - 1;
      float xv = (m >= 0 && m < 4096) ? xin[m >> 1] : 0.f;
      s = fmaf(xv, c0[(o * 3 + i) * 3 + tt], s);
    }
  }
  outA[idx] = s;
}

__global__ void convB_kernel(const float* __restrict__ outA,     // [16,6,4096]
                             const float* __restrict__ c1,       // [3,6,3]
                             float* __restrict__ outB) {         // [16,3,8192]
  long idx = (long)blockIdx.x * blockDim.x + threadIdx.x;
  if (idx >= (long)BATCH * 3 * 8192) return;
  int l = idx % 8192; int o = (idx / 8192) % 3; int b = idx / (8192 * 3);
  float s = 0.f;
  for (int i = 0; i < 6; ++i) {
    const float* xin = outA + ((long)b * 6 + i) * 4096;
    for (int tt = 0; tt < 3; ++tt) {
      int m = l + tt - 1;
      float xv = (m >= 0 && m < 8192) ? xin[m >> 1] : 0.f;
      s = fmaf(xv, c1[(o * 6 + i) * 3 + tt], s);
    }
  }
  outB[idx] = fmaxf(s, 0.f);
}

__global__ void convC_kernel(const float* __restrict__ outB,     // [16,3,8192]
                             const float* __restrict__ c2,       // [3,3,3]
                             float* __restrict__ outC) {         // [16,3,16384]
  long idx = (long)blockIdx.x * blockDim.x + threadIdx.x;
  if (idx >= (long)BATCH * 3 * 16384) return;
  int l = idx % 16384; int o = (idx / 16384) % 3; int b = idx / (16384 * 3);
  float s = 0.f;
  for (int i = 0; i < 3; ++i) {
    const float* xin = outB + ((long)b * 3 + i) * 8192;
    for (int tt = 0; tt < 3; ++tt) {
      int m = l + tt - 1;
      float xv = (m >= 0 && m < 16384) ? xin[m >> 1] : 0.f;
      s = fmaf(xv, c2[(o * 3 + i) * 3 + tt], s);
    }
  }
  outC[idx] = s;
}

// ===========================================================================
// 4. posn / res finalize (in-kernel select on device scalar d)
// ===========================================================================
__global__ void finalize_pos(const float* __restrict__ outC,
                             const float* __restrict__ pos,
                             const int* __restrict__ dflag,
                             float* __restrict__ dout) {
  long idx = (long)blockIdx.x * blockDim.x + threadIdx.x;
  if (idx >= (long)BATCH * NPTS) return;
  int n = idx % NPTS; int b = idx / NPTS;
  int dd = dflag[0];
  for (int ch = 0; ch < 3; ++ch) {
    float p  = pos[(long)n * 3 + ch];
    float rv = (dd > 0) ? outC[((long)b * 3 + ch) * 16384 + n] : p;
    float pn = (dd > 0) ? p + rv : p;
    dout[O_POSN + ((long)b * NPTS + n) * 3 + ch] = pn;
    dout[O_RES  + ((long)b * NPTS + n) * 3 + ch] = rv;
  }
}

// ===========================================================================
// 5. Trilinear splat (atomic scatter-add into V)
// ===========================================================================
__global__ void splat_kernel(const float* __restrict__ posn,
                             const float* __restrict__ amp,
                             float* __restrict__ vol) {
  long idx = (long)blockIdx.x * blockDim.x + threadIdx.x;
  if (idx >= (long)BATCH * NPTS) return;
  int b = idx / NPTS;
  const float a = amp[0];
  float i0f[3], fr[3];
  for (int d0 = 0; d0 < 3; ++d0) {
    float g = (posn[idx * 3 + d0] + 0.5f) * (float)BOX;
    float fl = floorf(g);
    i0f[d0] = fl;
    fr[d0] = g - fl;
  }
  float* vb = vol + (long)b * VOXB;
  for (int corner = 0; corner < 8; ++corner) {
    float w = a;
    int ci[3];
    for (int d0 = 0; d0 < 3; ++d0) {
      int bit = (corner >> (2 - d0)) & 1;
      w *= bit ? fr[d0] : (1.f - fr[d0]);
      float cf = fminf(fmaxf(i0f[d0] + (float)bit, 0.f), 127.f);
      ci[d0] = (int)cf;
    }
    long lin = ((long)ci[0] * BOX + ci[1]) * BOX + ci[2];
    atomicAdd(&vb[lin], w);
  }
}

// ===========================================================================
// 6. real -> complex copy
// ===========================================================================
__global__ void vol_to_complex(const float* __restrict__ vol, float2* __restrict__ vc) {
  long idx = (long)blockIdx.x * blockDim.x + threadIdx.x;
  if (idx >= N_VOL) return;
  vc[idx] = make_float2(vol[idx], 0.f);
}

// ===========================================================================
// 7. 128-pt centered FFT line kernel (one line / 128-thread block)
//    centered DFT (D even): Z[k] = (-1)^k * FFT[ (-1)^n x[n] ][k]
//    Contiguous passes (elem_stride==1) can stage the 1 KB line into LDS via
//    the Tensor Data Mover (tensor_load_to_lds + s_wait_tensorcnt).
// ===========================================================================
__global__ __launch_bounds__(128)
void fft128_kernel(float2* __restrict__ buf, long nlines,
                   long outer_stride, int inner_count, long inner_stride,
                   long elem_stride, int inverse, int use_tdm) {
  __shared__ float2 s[128];                        // at LDS offset 0
  const long line = blockIdx.x;
  if (line >= nlines) return;                      // uniform per block
  const long outer = line / inner_count;
  const long inner = line - outer * inner_count;
  const long base  = outer * outer_stride + inner * inner_stride;
  const int t = threadIdx.x;

  float2 v;
#if HAVE_TDM
  if (use_tdm) {
    if (t < 32) {                                  // wave 0 issues the DMA
      unsigned long long ga = (unsigned long long)(uintptr_t)(const void*)(buf + base);
      v4u g0;                                      // D# group 0 (ISA 8.3)
      g0[0] = 1u;                                  // count=1, user descriptor
      g0[1] = 0u;                                  // lds_addr = 0 (s[] base)
      g0[2] = (unsigned)(ga & 0xffffffffu);        // global_addr[31:0]
      g0[3] = (unsigned)(ga >> 32) | 0x80000000u;  // addr[56:32] | type=2
      v8i g1;                                      // D# group 1 (ISA 8.4)
      g1[0] = 0x20000;                             // data_size=2 (4 B dwords)
      g1[1] = (int)(256u << 16);                   // tensor_dim0 = 256 dwords
      g1[2] = (int)(1u << 16);                     // tensor_dim1 = 1
      g1[3] = (int)(256u << 16);                   // tile_dim0 = 256 dwords
      g1[4] = 0;                                   // tile_dim1/2 unused
      g1[5] = 256;                                 // tensor_dim0_stride
      g1[6] = 0;
      g1[7] = 0;
      v4i g2 = {0, 0, 0, 0}, g3 = {0, 0, 0, 0};
#if __clang_major__ >= 23
      v8i g4 = {0, 0, 0, 0, 0, 0, 0, 0};
      __builtin_amdgcn_tensor_load_to_lds(g0, g1, g2, g3, g4, 0);
#else
      __builtin_amdgcn_tensor_load_to_lds(g0, g1, g2, g3, 0);
#endif
      __builtin_amdgcn_s_wait_tensorcnt(0);
    }
    __syncthreads();                               // LDS now holds the line
    v = s[t];
    __syncthreads();                               // before bit-rev overwrite
  } else
#endif
  {
    (void)use_tdm;
    v = buf[base + (long)t * elem_stride];
  }

  if (t & 1) { v.x = -v.x; v.y = -v.y; }           // pre-phase (-1)^n
  const int rt = (int)(__brev((unsigned)t) >> 25); // 7-bit reversal
  s[rt] = v;
  __syncthreads();

  const float sign = inverse ? 1.f : -1.f;
  for (int stage = 1; stage <= 7; ++stage) {
    const int len = 1 << stage, half = len >> 1;
    if (t < 64) {
      const int j  = t & (half - 1);
      const int g  = t >> (stage - 1);
      const int i0 = g * len + j;
      const int i1 = i0 + half;
      const float ang = sign * 6.2831853071795864f * (float)j / (float)len;
      float sw, cw;
      __sincosf(ang, &sw, &cw);
      float2 u = s[i0], x = s[i1];
      float2 wv = make_float2(x.x * cw - x.y * sw, x.x * sw + x.y * cw);
      s[i0] = make_float2(u.x + wv.x, u.y + wv.y);
      s[i1] = make_float2(u.x - wv.x, u.y - wv.y);
    }
    __syncthreads();
  }

  float2 o = s[t];
  if (inverse) { o.x *= (1.f / 128.f); o.y *= (1.f / 128.f); }
  if (t & 1) { o.x = -o.x; o.y = -o.y; }           // post-phase (-1)^k
  buf[base + (long)t * elem_stride] = o;
}

// ===========================================================================
// 8. Fourier-slice trilinear gather
// ===========================================================================
__global__ void slice_gather(const float2* __restrict__ vc,
                             const float* __restrict__ r,
                             float2* __restrict__ sl) {
  long idx = (long)blockIdx.x * blockDim.x + threadIdx.x;
  if (idx >= (long)BATCH * BOX * BOX) return;
  int j = idx % BOX; int i = (idx / BOX) % BOX; int b = idx / (BOX * BOX);
  const float kx = (float)i - 64.f, ky = (float)j - 64.f;
  const float* rb = r + (long)b * 9;
  float i0f[3], fr[3];
  for (int c = 0; c < 3; ++c) {
    float crd = rb[c * 3 + 0] * kx + rb[c * 3 + 1] * ky + 64.f;
    float fl = floorf(crd);
    i0f[c] = fl;
    fr[c] = crd - fl;
  }
  const float2* vb = vc + (long)b * VOXB;
  float sx = 0.f, sy = 0.f;
  for (int corner = 0; corner < 8; ++corner) {
    float w = 1.f;
    int ci[3];
    for (int c = 0; c < 3; ++c) {
      int bit = (corner >> (2 - c)) & 1;
      w *= bit ? fr[c] : (1.f - fr[c]);
      float cf = fminf(fmaxf(i0f[c] + (float)bit, 0.f), 127.f);
      ci[c] = (int)cf;
    }
    float2 v = vb[((long)ci[0] * BOX + ci[1]) * BOX + ci[2]];
    sx = fmaf(v.x, w, sx);
    sy = fmaf(v.y, w, sy);
  }
  sl[idx] = make_float2(sx, sy);
}

// ===========================================================================
// 9. Real-part extraction and W broadcast
// ===========================================================================
__global__ void extract_real(const float2* __restrict__ sl, float* __restrict__ proj) {
  long idx = (long)blockIdx.x * blockDim.x + threadIdx.x;
  if (idx >= N_PROJ) return;
  proj[idx] = sl[idx].x;
}

__global__ void wb_bcast(const float* __restrict__ W, float* __restrict__ wb) {
  long idx = (long)blockIdx.x * blockDim.x + threadIdx.x;
  if (idx >= N_WB) return;
  wb[idx] = W[idx % (96 * 96)];
}

// ===========================================================================
// launch
// ===========================================================================
extern "C" void kernel_launch(void* const* d_in, const int* in_sizes, int n_in,
                              void* d_out, int out_size, void* d_ws, size_t ws_size,
                              hipStream_t stream) {
  const float* z     = (const float*)d_in[0];
  const float* r     = (const float*)d_in[1];
  const int*   dflag = (const int*)d_in[2];
  const float* pos   = (const float*)d_in[3];
  const float* W     = (const float*)d_in[4];
  const float* amp   = (const float*)d_in[5];
  const float* w1    = (const float*)d_in[6];
  const float* w2    = (const float*)d_in[7];
  const float* w3    = (const float*)d_in[8];
  const float* w4    = (const float*)d_in[9];
  const float* w5    = (const float*)d_in[10];
  const float* w6    = (const float*)d_in[11];
  const float* c0    = (const float*)d_in[12];
  const float* c1    = (const float*)d_in[13];
  const float* c2    = (const float*)d_in[14];

  float* out = (float*)d_out;
  char*  ws  = (char*)d_ws;
  _Float16* a16f = (_Float16*)(ws + WS_A16F);
  float*    res0 = (float*)(ws + WS_RES0);
  float*    oA   = (float*)(ws + WS_C0);
  float*    oB   = (float*)(ws + WS_C1);
  float*    oC   = (float*)(ws + WS_C2);
  float2*   sl   = (float2*)(ws + WS_SLICE);
  _Float16* w6f  = (_Float16*)(ws + WS_W6F);
  float2*   vc   = (float2*)(ws + WS_VC);

  // V accumulates via atomics -> must be zeroed every call (capturable memset)
  hipMemsetAsync(out + O_VOL, 0, (size_t)N_VOL * sizeof(float), stream);

  // MLP + WMMA GEMM + conv chain
  mlp_head<<<1, 256, 0, stream>>>(z, w1, w2, w3, w4, w5, a16f);
  w6_to_frag<<<((long)NTILE * NKT * 512 + 255) / 256, 256, 0, stream>>>(w6, w6f);
  gemm_w6_wmma<<<96, 128, 0, stream>>>(a16f, w6f, res0);
  convA_kernel<<<((long)BATCH * 6 * 4096 + 255) / 256, 256, 0, stream>>>(res0, c0, oA);
  convB_kernel<<<((long)BATCH * 3 * 8192 + 255) / 256, 256, 0, stream>>>(oA, c1, oB);
  convC_kernel<<<((long)BATCH * 3 * 16384 + 255) / 256, 256, 0, stream>>>(oB, c2, oC);
  finalize_pos<<<((long)BATCH * NPTS + 255) / 256, 256, 0, stream>>>(oC, pos, dflag, out);

  // splat into V (output region), then real->complex
  splat_kernel<<<((long)BATCH * NPTS + 255) / 256, 256, 0, stream>>>(
      out + O_POSN, amp, out + O_VOL);
  vol_to_complex<<<(N_VOL + 255) / 256, 256, 0, stream>>>(out + O_VOL, vc);

  // forward centered 3D FFT: z (TDM-staged), y, x passes
  fft128_kernel<<<262144, 128, 0, stream>>>(vc, 262144L, 128L,      1,     0L, 1L,     0, 1);
  fft128_kernel<<<262144, 128, 0, stream>>>(vc, 262144L, 16384L,    128,   1L, 128L,   0, 0);
  fft128_kernel<<<262144, 128, 0, stream>>>(vc, 262144L, 2097152L,  16384, 1L, 16384L, 0, 0);

  // Fourier-slice gather + centered 2D inverse FFT (scale 1/128 per pass)
  slice_gather<<<((long)BATCH * BOX * BOX + 255) / 256, 256, 0, stream>>>(vc, r, sl);
  fft128_kernel<<<2048, 128, 0, stream>>>(sl, 2048L, 128L,   1,   0L, 1L,   1, 1);
  fft128_kernel<<<2048, 128, 0, stream>>>(sl, 2048L, 16384L, 128, 1L, 128L, 1, 0);

  extract_real<<<(N_PROJ + 255) / 256, 256, 0, stream>>>(sl, out + O_PROJ);
  wb_bcast<<<(N_WB + 255) / 256, 256, 0, stream>>>(W, out + O_WB);
}
